// Decoder_85229331022208
// MI455X (gfx1250) — compile-verified, matrix operand order checked by reading, not run on hardware
//
#include <hip/hip_runtime.h>
#include <hip/hip_bf16.h>
#include <math.h>

// ---------------------------------------------------------------------------
// Types for CDNA5 WMMA (gfx1250, wave32)
// ---------------------------------------------------------------------------
typedef __bf16        v16bf  __attribute__((ext_vector_type(16)));
typedef float         v8f    __attribute__((ext_vector_type(8)));
typedef unsigned int  uint4v __attribute__((ext_vector_type(4)));
typedef unsigned int  tdm_u32x4 __attribute__((ext_vector_type(4)));
typedef int           tdm_i32x4 __attribute__((ext_vector_type(4)));
typedef int           tdm_i32x8 __attribute__((ext_vector_type(8)));

#if defined(__AMDGCN__) && __has_builtin(__builtin_amdgcn_tensor_load_to_lds) && \
    __has_builtin(__builtin_amdgcn_s_wait_tensorcnt)
#define HAVE_TDM 1
#else
#define HAVE_TDM 0
#endif

__device__ inline v8f wmma_bf16(v16bf a, v16bf b, v8f c) {
  // D = A(16x32 bf16) x B(32x16 bf16) + C(16x16 f32)
  return __builtin_amdgcn_wmma_f32_16x16x32_bf16(
      false, a, false, b, (short)0, c, false, false);
}

// fp32 -> bf16 with round-to-nearest-even
__device__ inline unsigned short f2bf(float f) {
  unsigned int u = __float_as_uint(f);
  unsigned int r = (u + 0x7FFFu + ((u >> 16) & 1u)) >> 16;
  return (unsigned short)r;
}

// Load a 16-lane x 32-K bf16 fragment from LDS (row-major [rows][ld=40] halfs).
// Per-lane ISA layout for 16-bit A (mirrored for B): lane L holds row (L&15),
// K = kb+0..7 in elements 0..7 and K = kb+16..23 in elements 8..15, kb=(L>>4)*8.
__device__ inline v16bf ldfrag(const unsigned short* p, int kb) {
  union { v16bf v; uint4v q[2]; } u;
  u.q[0] = *(const uint4v*)(p + kb);
  u.q[1] = *(const uint4v*)(p + kb + 16);
  return u.v;
}

#if HAVE_TDM
// ---------------------------------------------------------------------------
// TDM: DMA one 2D bf16 tile (tile_rows x 32 elements) from global to LDS.
// D# built per CDNA5 ISA 8.3/8.4. Pad feature inserts 4 DWORDs (16B) after
// every 16 DWORDs (64B = one 32-half row) -> LDS row stride of 40 halfs,
// matching the bank-conflict-free layout the WMMA fragment loader expects.
// ---------------------------------------------------------------------------
__device__ inline void tdm_load_tile32(unsigned int lds_off,
                                       const unsigned short* gtile,
                                       int tensor_w, int tensor_h,
                                       int tile_rows) {
  unsigned long long ga = (unsigned long long)gtile;
  tdm_u32x4 g0;
  g0[0] = 1u;                                          // count=1 (valid), user mode
  g0[1] = lds_off;                                     // lds_addr (bytes)
  g0[2] = (unsigned int)(ga & 0xFFFFFFFFu);            // global_addr[31:0]
  g0[3] = (unsigned int)((ga >> 32) & 0x01FFFFFFu) | (2u << 30);  // [56:32] | type=2
  tdm_i32x8 g1;
  g1[0] = (int)((1u << 16) |        // data_size = 1 (2 bytes)
                (1u << 20) |        // pad_enable
                (3u << 22) |        // pad_interval: 16 DWORDs
                (3u << 25));        // pad_amount:   4 DWORDs
  g1[1] = (int)(((unsigned)tensor_w & 0xFFFFu) << 16);              // dim0[15:0]
  g1[2] = (int)((((unsigned)tensor_w >> 16) & 0xFFFFu) |
                (((unsigned)tensor_h & 0xFFFFu) << 16));            // dim0[31:16]|dim1[15:0]
  g1[3] = (int)((((unsigned)tensor_h >> 16) & 0xFFFFu) | (32u << 16)); // dim1[31:16]|tile_dim0=32
  g1[4] = tile_rows;                                                // tile_dim1
  g1[5] = tensor_w;                                                 // dim0_stride[31:0]
  g1[6] = 0;
  g1[7] = 0;
  tdm_i32x4 z4 = {0, 0, 0, 0};
#if defined(__clang_major__) && __clang_major__ >= 23
  tdm_i32x8 z8 = {0, 0, 0, 0, 0, 0, 0, 0};
  __builtin_amdgcn_tensor_load_to_lds(g0, g1, z4, z4, z8, 0);
#else
  __builtin_amdgcn_tensor_load_to_lds(g0, g1, z4, z4, 0);
#endif
}
#endif  // HAVE_TDM

// ---------------------------------------------------------------------------
// Weight conversion kernels
// ---------------------------------------------------------------------------
__global__ void cvt_bf16_kernel(const float* __restrict__ w,
                                unsigned short* __restrict__ o, int n) {
  int i = blockIdx.x * 256 + threadIdx.x;
  if (i < n) o[i] = f2bf(w[i]);
}

__global__ void cvt_bf16_t_kernel(const float* __restrict__ w,
                                  unsigned short* __restrict__ o, int R, int C) {
  int i = blockIdx.x * 256 + threadIdx.x;
  if (i < R * C) {
    int r = i / C, c = i - r * C;
    o[(size_t)c * R + r] = f2bf(w[i]);   // o = w^T
  }
}

// ---------------------------------------------------------------------------
// LayerNorm over last dim (D=1024): fp32 in -> bf16 out.  One block per row.
// ---------------------------------------------------------------------------
__global__ __launch_bounds__(256)
void ln_kernel(const float* __restrict__ x, const float* __restrict__ g,
               const float* __restrict__ b, unsigned short* __restrict__ o,
               int D) {
  int row = blockIdx.x, tid = threadIdx.x;
  const float* xr = x + (size_t)row * D;
  float vals[4], s = 0.f, ss = 0.f;
#pragma unroll
  for (int i = 0; i < 4; ++i) {
    float v = xr[tid + i * 256];
    vals[i] = v; s += v; ss += v * v;
  }
  __shared__ float r1[256], r2[256];
  r1[tid] = s; r2[tid] = ss; __syncthreads();
  for (int off = 128; off; off >>= 1) {
    if (tid < off) { r1[tid] += r1[tid + off]; r2[tid] += r2[tid + off]; }
    __syncthreads();
  }
  float mean = r1[0] / (float)D;
  float var  = r2[0] / (float)D - mean * mean;
  float rstd = rsqrtf(var + 1e-6f);
  unsigned short* orow = o + (size_t)row * D;
#pragma unroll
  for (int i = 0; i < 4; ++i) {
    int d = tid + i * 256;
    orow[d] = f2bf((vals[i] - mean) * rstd * g[d] + b[d]);
  }
}

// ---------------------------------------------------------------------------
// Row L2 norm: tvnorm[row] = ||tv[row,:]||
// ---------------------------------------------------------------------------
__global__ __launch_bounds__(256)
void rownorm_kernel(const float* __restrict__ x, float* __restrict__ o, int D) {
  int row = blockIdx.x, tid = threadIdx.x;
  const float* xr = x + (size_t)row * D;
  float s = 0.f;
  for (int i = tid; i < D; i += 256) { float v = xr[i]; s += v * v; }
  __shared__ float red[256];
  red[tid] = s; __syncthreads();
  for (int off = 128; off; off >>= 1) {
    if (tid < off) red[tid] += red[tid + off];
    __syncthreads();
  }
  if (tid == 0) o[row] = sqrtf(red[0]);
}

// ---------------------------------------------------------------------------
// GEMM: C[M,N] = A[M,K](bf16) @ W[N,K]^T(bf16) + bias [+gelu] [+resid]
// Block tile 128x128, K-step 32. 256 threads = 8 waves in a 4x2 grid; each
// wave owns a 32x64 C tile = 2x4 WMMA 16x16x32 tiles (8 WMMA / k-step).
// Tiles are staged via the Tensor Data Mover into double-buffered LDS
// (s_wait_tensorcnt + barrier hand-off); falls back to manual staging if the
// TDM builtin is unavailable.
// ---------------------------------------------------------------------------
__global__ __launch_bounds__(256)
void gemm_bt_kernel(const unsigned short* __restrict__ A,
                    const unsigned short* __restrict__ W,
                    const float* __restrict__ bias,
                    const float* __restrict__ resid,
                    float* __restrict__ outf,
                    unsigned short* __restrict__ outb,
                    int M, int N, int K, int act_gelu) {
#if HAVE_TDM
  __shared__ unsigned short As[2][128 * 40];
  __shared__ unsigned short Bs[2][128 * 40];
#else
  __shared__ unsigned short As[1][128 * 40];
  __shared__ unsigned short Bs[1][128 * 40];
#endif
  const int tid  = threadIdx.x;
  const int lane = tid & 31, wave = tid >> 5;
  const int wm = wave >> 1, wn = wave & 1;      // 4 x 2 wave grid
  const int fr  = lane & 15;
  const int kb  = (lane >> 4) * 8;              // K fragment base (halfs)
  const int hi8 = (lane >> 4) * 8;              // C-tile row base
  const size_t m0 = (size_t)blockIdx.y * 128;
  const int    n0 = blockIdx.x * 128;
  const int    nk = K / 32;

  v8f acc[2][4] = {};

#if HAVE_TDM
  if (wave == 0) {
    tdm_load_tile32((unsigned int)(unsigned long long)&As[0][0],
                    &A[m0 * (size_t)K], K, M, 128);
    tdm_load_tile32((unsigned int)(unsigned long long)&Bs[0][0],
                    &W[(size_t)n0 * K], K, N, 128);
  }
#endif

  for (int kt = 0; kt < nk; ++kt) {
    const int kk = kt * 32;
#if HAVE_TDM
    const int cur = kt & 1;
    if (wave == 0) {
      if (kt + 1 < nk) {
        tdm_load_tile32((unsigned int)(unsigned long long)&As[cur ^ 1][0],
                        &A[m0 * (size_t)K + kk + 32], K, M, 128);
        tdm_load_tile32((unsigned int)(unsigned long long)&Bs[cur ^ 1][0],
                        &W[(size_t)n0 * K + kk + 32], K, N, 128);
        __builtin_amdgcn_s_wait_tensorcnt(2);   // current buffer's pair done
      } else {
        __builtin_amdgcn_s_wait_tensorcnt(0);
      }
    }
    __syncthreads();   // publish TDM-written LDS to all waves
#else
    const int cur = 0;
    for (int i = tid; i < 512; i += 256) {
      int r = i >> 2, c = i & 3;
      *(uint4v*)&As[0][r * 40 + c * 8] =
          *(const uint4v*)&A[(m0 + r) * (size_t)K + kk + c * 8];
      *(uint4v*)&Bs[0][r * 40 + c * 8] =
          *(const uint4v*)&W[(size_t)(n0 + r) * K + kk + c * 8];
    }
    if (kk + 32 < K) {
      __builtin_prefetch(&A[(m0 + (tid >> 1)) * (size_t)K + kk + 32], 0, 0);
      __builtin_prefetch(&W[(size_t)(n0 + (tid >> 1)) * K + kk + 32], 0, 0);
    }
    __syncthreads();
#endif

    v16bf a0 = ldfrag(&As[cur][(wm * 32 +      fr) * 40], kb);
    v16bf a1 = ldfrag(&As[cur][(wm * 32 + 16 + fr) * 40], kb);
    v16bf b0 = ldfrag(&Bs[cur][(wn * 64 +      fr) * 40], kb);
    v16bf b1 = ldfrag(&Bs[cur][(wn * 64 + 16 + fr) * 40], kb);
    v16bf b2 = ldfrag(&Bs[cur][(wn * 64 + 32 + fr) * 40], kb);
    v16bf b3 = ldfrag(&Bs[cur][(wn * 64 + 48 + fr) * 40], kb);
    acc[0][0] = wmma_bf16(a0, b0, acc[0][0]);
    acc[0][1] = wmma_bf16(a0, b1, acc[0][1]);
    acc[0][2] = wmma_bf16(a0, b2, acc[0][2]);
    acc[0][3] = wmma_bf16(a0, b3, acc[0][3]);
    acc[1][0] = wmma_bf16(a1, b0, acc[1][0]);
    acc[1][1] = wmma_bf16(a1, b1, acc[1][1]);
    acc[1][2] = wmma_bf16(a1, b2, acc[1][2]);
    acc[1][3] = wmma_bf16(a1, b3, acc[1][3]);
    __syncthreads();   // all reads of this buffer done before it is refilled
  }

#pragma unroll
  for (int tm = 0; tm < 2; ++tm)
#pragma unroll
    for (int tn = 0; tn < 4; ++tn)
#pragma unroll
      for (int v = 0; v < 8; ++v) {
        size_t gm = m0 + wm * 32 + tm * 16 + hi8 + v;
        int    gn = n0 + wn * 64 + tn * 16 + fr;
        float c = acc[tm][tn][v] + (bias ? bias[gn] : 0.f);
        if (act_gelu) c = 0.5f * c * (1.f + erff(c * 0.70710678118f));
        if (resid) c += resid[gm * (size_t)N + gn];
        if (outf)  outf[gm * (size_t)N + gn] = c;
        if (outb)  outb[gm * (size_t)N + gn] = f2bf(c);
      }
}

// ---------------------------------------------------------------------------
// Flash attention, hd=64. One block = one (b,h,32-query tile). 128 thr/4 waves.
// Scores = Q.K^T/8 (WMMA), online softmax, O += P.V (WMMA). Optional causal
// mask and optional (m,l) stats output (for the wvn recompute pass).
// Output bf16 at [b*T+t, h*64 + d] (heads concatenated).
// ---------------------------------------------------------------------------
__global__ __launch_bounds__(128)
void flash_kernel(const unsigned short* __restrict__ Qp, int ldq, int qcol0,
                  const unsigned short* __restrict__ Kp, int ldk, int kcol0,
                  const unsigned short* __restrict__ Vp, int ldv, int vcol0,
                  unsigned short* __restrict__ Op, int ldo,
                  float* __restrict__ m_out, float* __restrict__ l_out,
                  int B, int H, int T, int S, int causal) {
  __shared__ unsigned short Qs[32 * 72];   // 32 x 64 (+8 pad)
  __shared__ unsigned short Ks[32 * 72];
  __shared__ unsigned short Vt[64 * 40];   // V transposed: [d][key] 64x32(+8)
  __shared__ float          Ssm[32 * 36];  // fp32 scores
  __shared__ unsigned short Ps[32 * 40];   // bf16 probs
  __shared__ float mrow[32], lrow[32], crow[32];

  const int tid = threadIdx.x;
  const int nqt = T / 32;
  const int qt  = blockIdx.x % nqt;
  const int h   = (blockIdx.x / nqt) % H;
  const int b   = blockIdx.x / (nqt * H);
  const int qcol = qcol0 + h * 64, kcol = kcol0 + h * 64, vcol = vcol0 + h * 64;

  const int lane = tid & 31, wave = tid >> 5;
  const int mt = wave & 1, st = wave >> 1;
  const int fr = lane & 15, kb = (lane >> 4) * 8, hi8 = (lane >> 4) * 8;

  for (int i = tid; i < 256; i += 128) {
    int r = i >> 3, c = i & 7;
    *(uint4v*)&Qs[r * 72 + c * 8] =
        *(const uint4v*)&Qp[((size_t)(b * T + qt * 32 + r)) * ldq + qcol + c * 8];
  }
  if (tid < 32) { mrow[tid] = -1e30f; lrow[tid] = 0.f; }
  v8f o0 = {}, o1 = {};
  __syncthreads();

  const int ktend = causal ? qt : (S / 32 - 1);
  for (int kt = 0; kt <= ktend; ++kt) {
    for (int i = tid; i < 256; i += 128) {
      int r = i >> 3, c = i & 7;
      *(uint4v*)&Ks[r * 72 + c * 8] =
          *(const uint4v*)&Kp[((size_t)(b * S + kt * 32 + r)) * ldk + kcol + c * 8];
    }
    for (int i = tid; i < 256; i += 128) {
      int r = i >> 3, c = i & 7;
      union { uint4v q; unsigned short u[8]; } uu;
      uu.q = *(const uint4v*)&Vp[((size_t)(b * S + kt * 32 + r)) * ldv + vcol + c * 8];
#pragma unroll
      for (int j = 0; j < 8; ++j) Vt[(c * 8 + j) * 40 + r] = uu.u[j];
    }
    __syncthreads();

    v8f sc = {};
    {
      v16bf qa = ldfrag(&Qs[(mt * 16 + fr) * 72], kb);
      v16bf ka = ldfrag(&Ks[(st * 16 + fr) * 72], kb);
      sc = wmma_bf16(qa, ka, sc);
      qa = ldfrag(&Qs[(mt * 16 + fr) * 72 + 32], kb);
      ka = ldfrag(&Ks[(st * 16 + fr) * 72 + 32], kb);
      sc = wmma_bf16(qa, ka, sc);
    }
#pragma unroll
    for (int v = 0; v < 8; ++v) {
      int ml = mt * 16 + hi8 + v;
      int nl = st * 16 + fr;
      float s = sc[v] * 0.125f;                       // 1/sqrt(64)
      if (causal) {
        int tq = qt * 32 + ml, tk = kt * 32 + nl;
        if (tk > tq) s = -1e30f;
      }
      Ssm[ml * 36 + nl] = s;
    }
    __syncthreads();

    if (tid < 32) {
      float mo = mrow[tid], mx = mo;
      for (int j = 0; j < 32; ++j) mx = fmaxf(mx, Ssm[tid * 36 + j]);
      float corr = __expf(mo - mx), ps = 0.f;
      for (int j = 0; j < 32; ++j) {
        float p = __expf(Ssm[tid * 36 + j] - mx);
        Ps[tid * 40 + j] = f2bf(p);
        ps += p;
      }
      mrow[tid] = mx; lrow[tid] = lrow[tid] * corr + ps; crow[tid] = corr;
    }
    __syncthreads();

    {
      v16bf pa = ldfrag(&Ps[(mt * 16 + fr) * 40], kb);
      v16bf v0 = ldfrag(&Vt[(st * 32 +      fr) * 40], kb);
      v16bf v1 = ldfrag(&Vt[(st * 32 + 16 + fr) * 40], kb);
#pragma unroll
      for (int v = 0; v < 8; ++v) {
        float cr = crow[mt * 16 + hi8 + v];
        o0[v] *= cr; o1[v] *= cr;
      }
      o0 = wmma_bf16(pa, v0, o0);
      o1 = wmma_bf16(pa, v1, o1);
    }
    __syncthreads();
  }

#pragma unroll
  for (int v = 0; v < 8; ++v) {
    int ml = mt * 16 + hi8 + v;
    float inv = 1.0f / lrow[ml];
    size_t row = (size_t)(b * T + qt * 32 + ml);
    Op[row * ldo + h * 64 + st * 32 +      fr] = f2bf(o0[v] * inv);
    Op[row * ldo + h * 64 + st * 32 + 16 + fr] = f2bf(o1[v] * inv);
  }
  if (m_out && tid < 32) {
    int idx = (b * H + h) * T + qt * 32 + tid;
    m_out[idx] = mrow[tid];
    l_out[idx] = lrow[tid];
  }
}

// ---------------------------------------------------------------------------
// wvn[b,t,s] = (1/H) sum_h softmax_h(q.k/8)[t,s] * ||tv[b,s,:]||
// Recomputes scores per head via WMMA using saved (m,l).
// ---------------------------------------------------------------------------
__global__ __launch_bounds__(128)
void wvn_kernel(const unsigned short* __restrict__ Q2, int ldq,
                const unsigned short* __restrict__ K2, int ldk, int kcol0,
                const float* __restrict__ m_buf, const float* __restrict__ l_buf,
                const float* __restrict__ tvnorm, float* __restrict__ wvn,
                int B, int H, int T, int S) {
  __shared__ unsigned short Qs[32 * 72];
  __shared__ unsigned short Ks[32 * 72];
  __shared__ float mls[32], ils[32];

  const int tid = threadIdx.x;
  const int ntt = T / 32;
  const int t0  = (blockIdx.x % ntt) * 32;
  const int b   = blockIdx.x / ntt;
  const int lane = tid & 31, wave = tid >> 5;
  const int mt = wave & 1, nt = wave >> 1;
  const int fr = lane & 15, kb = (lane >> 4) * 8, hi8 = (lane >> 4) * 8;
  const float invH = 1.0f / (float)H;

  for (int s0 = 0; s0 < S; s0 += 32) {
    v8f acc = {};
    for (int h = 0; h < H; ++h) {
      __syncthreads();
      for (int i = tid; i < 256; i += 128) {
        int r = i >> 3, c = i & 7;
        *(uint4v*)&Qs[r * 72 + c * 8] =
            *(const uint4v*)&Q2[((size_t)(b * T + t0 + r)) * ldq + h * 64 + c * 8];
        *(uint4v*)&Ks[r * 72 + c * 8] =
            *(const uint4v*)&K2[((size_t)(b * S + s0 + r)) * ldk + kcol0 + h * 64 + c * 8];
      }
      if (tid < 32) {
        int idx = (b * H + h) * T + t0 + tid;
        mls[tid] = m_buf[idx];
        ils[tid] = 1.0f / l_buf[idx];
      }
      __syncthreads();

      v8f sc = {};
      v16bf qa = ldfrag(&Qs[(mt * 16 + fr) * 72], kb);
      v16bf ka = ldfrag(&Ks[(nt * 16 + fr) * 72], kb);
      sc = wmma_bf16(qa, ka, sc);
      qa = ldfrag(&Qs[(mt * 16 + fr) * 72 + 32], kb);
      ka = ldfrag(&Ks[(nt * 16 + fr) * 72 + 32], kb);
      sc = wmma_bf16(qa, ka, sc);

#pragma unroll
      for (int v = 0; v < 8; ++v) {
        int tl = mt * 16 + hi8 + v;
        acc[v] += __expf(sc[v] * 0.125f - mls[tl]) * ils[tl] * invH;
      }
    }
#pragma unroll
    for (int v = 0; v < 8; ++v) {
      int tl = mt * 16 + hi8 + v;
      int gs = s0 + nt * 16 + fr;
      wvn[((size_t)(b * T + t0 + tl)) * S + gs] = acc[v] * tvnorm[b * S + gs];
    }
  }
}

// ---------------------------------------------------------------------------
// Host orchestration
// ---------------------------------------------------------------------------
extern "C" void kernel_launch(void* const* d_in, const int* in_sizes, int n_in,
                              void* d_out, int out_size, void* d_ws, size_t ws_size,
                              hipStream_t stream) {
  (void)in_sizes; (void)n_in; (void)out_size; (void)ws_size;
  const int B = 4, T = 1024, S = 1024, D = 1024, H = 16;
  const int M = B * T;  // 4096 rows for all row-major GEMMs

  const float* dec    = (const float*)d_in[0];
  const float* enc    = (const float*)d_in[1];
  const float* ln_g   = (const float*)d_in[4];
  const float* ln_b   = (const float*)d_in[5];
  const float* w_in1  = (const float*)d_in[6];
  const float* b_in1  = (const float*)d_in[7];
  const float* w_out1 = (const float*)d_in[8];
  const float* b_out1 = (const float*)d_in[9];
  const float* w_in2  = (const float*)d_in[10];
  const float* b_in2  = (const float*)d_in[11];
  const float* w_out2 = (const float*)d_in[12];
  const float* b_out2 = (const float*)d_in[13];
  const float* mlp_w1 = (const float*)d_in[14];
  const float* mlp_b1 = (const float*)d_in[15];
  const float* mlp_w2 = (const float*)d_in[16];
  const float* mlp_b2 = (const float*)d_in[17];

  float* out_x   = (float*)d_out;                      // [B,T,D]
  float* out_wvn = (float*)d_out + (size_t)B * T * D;  // [B,T,S]

  char* ws = (char*)d_ws;
  size_t off = 0;
  auto alloc = [&](size_t bytes) -> char* {
    char* p = ws + off;
    off = (off + bytes + 255) & ~(size_t)255;
    return p;
  };
  unsigned short* w_in1b   = (unsigned short*)alloc((size_t)3 * D * D * 2);
  unsigned short* w_out1b  = (unsigned short*)alloc((size_t)D * D * 2);
  unsigned short* w_in2b   = (unsigned short*)alloc((size_t)3 * D * D * 2);
  unsigned short* w_out2b  = (unsigned short*)alloc((size_t)D * D * 2);
  unsigned short* w_out2tb = (unsigned short*)alloc((size_t)D * D * 2);
  unsigned short* mlp_w1b  = (unsigned short*)alloc((size_t)4 * D * D * 2);
  unsigned short* mlp_w2b  = (unsigned short*)alloc((size_t)4 * D * D * 2);
  unsigned short* de_b     = (unsigned short*)alloc((size_t)M * D * 2);
  unsigned short* qkv1     = (unsigned short*)alloc((size_t)M * 3 * D * 2);
  unsigned short* attn1    = (unsigned short*)alloc((size_t)M * D * 2);
  float*          x_f32    = (float*)alloc((size_t)M * D * 4);
  unsigned short* x_b      = (unsigned short*)alloc((size_t)M * D * 2);
  unsigned short* en_b     = (unsigned short*)alloc((size_t)M * D * 2);
  float*          x2_f32   = (float*)alloc((size_t)M * D * 4);
  float*          m_buf    = (float*)alloc((size_t)B * H * T * 4);
  float*          l_buf    = (float*)alloc((size_t)B * H * T * 4);
  float*          tvnorm   = (float*)alloc((size_t)B * S * 4);
  unsigned short* h_b      = (unsigned short*)alloc((size_t)M * 4 * D * 2);
  // safe sequential aliases
  unsigned short* q2    = qkv1;                       // qkv1 dead after self-attn out-proj
  unsigned short* kv2   = qkv1 + (size_t)M * D;
  unsigned short* attn2 = de_b;                       // de_b dead after QKV GEMM
  float*          tv    = x_f32;                      // x dead after cross out-proj
  unsigned short* lnx_b = attn1;                      // attn1 dead after out-proj1

  auto cg = [](size_t n) { return dim3((unsigned)((n + 255) / 256)); };

  // 1) weights -> bf16
  cvt_bf16_kernel<<<cg((size_t)3*D*D), 256, 0, stream>>>(w_in1,  w_in1b,  3*D*D);
  cvt_bf16_kernel<<<cg((size_t)D*D),   256, 0, stream>>>(w_out1, w_out1b, D*D);
  cvt_bf16_kernel<<<cg((size_t)3*D*D), 256, 0, stream>>>(w_in2,  w_in2b,  3*D*D);
  cvt_bf16_kernel<<<cg((size_t)D*D),   256, 0, stream>>>(w_out2, w_out2b, D*D);
  cvt_bf16_t_kernel<<<cg((size_t)D*D), 256, 0, stream>>>(w_out2, w_out2tb, D, D);
  cvt_bf16_kernel<<<cg((size_t)4*D*D), 256, 0, stream>>>(mlp_w1, mlp_w1b, 4*D*D);
  cvt_bf16_kernel<<<cg((size_t)4*D*D), 256, 0, stream>>>(mlp_w2, mlp_w2b, 4*D*D);

  // 2) de_in = LN(decoder_input); qkv1 = de_in @ w_in1^T + b_in1
  ln_kernel<<<M, 256, 0, stream>>>(dec, ln_g, ln_b, de_b, D);
  gemm_bt_kernel<<<dim3(3*D/128, M/128), 256, 0, stream>>>(
      de_b, w_in1b, b_in1, nullptr, nullptr, qkv1, M, 3*D, D, 0);

  // 3) causal self-attention -> attn1
  flash_kernel<<<B*H*(T/32), 128, 0, stream>>>(
      qkv1, 3*D, 0, qkv1, 3*D, D, qkv1, 3*D, 2*D,
      attn1, D, nullptr, nullptr, B, H, T, S, 1);

  // 4) x = decoder_input + attn1 @ w_out1^T + b_out1  (fp32 + bf16 copies)
  gemm_bt_kernel<<<dim3(D/128, M/128), 256, 0, stream>>>(
      attn1, w_out1b, b_out1, dec, x_f32, x_b, M, D, D, 0);

  // 5) en_out = LN(encoder_output); cross QKV
  ln_kernel<<<M, 256, 0, stream>>>(enc, ln_g, ln_b, en_b, D);
  gemm_bt_kernel<<<dim3(D/128, M/128), 256, 0, stream>>>(
      x_b, w_in2b, b_in2, nullptr, nullptr, q2, M, D, D, 0);
  gemm_bt_kernel<<<dim3(2*D/128, M/128), 256, 0, stream>>>(
      en_b, w_in2b + (size_t)D * D, b_in2 + D, nullptr, nullptr, kv2, M, 2*D, D, 0);

  // 6) cross-attention -> attn2, saving (m,l) for the wvn recompute pass
  flash_kernel<<<B*H*(T/32), 128, 0, stream>>>(
      q2, D, 0, kv2, 2*D, 0, kv2, 2*D, D,
      attn2, D, m_buf, l_buf, B, H, T, S, 0);

  // 7) x2 = x + attn2 @ w_out2^T + b_out2
  gemm_bt_kernel<<<dim3(D/128, M/128), 256, 0, stream>>>(
      attn2, w_out2b, b_out2, x_f32, x2_f32, nullptr, M, D, D, 0);

  // 8) tv = en_out @ w_out2 (untransposed) ; tvnorm = ||tv||_row
  gemm_bt_kernel<<<dim3(D/128, M/128), 256, 0, stream>>>(
      en_b, w_out2tb, nullptr, nullptr, tv, nullptr, M, D, D, 0);
  rownorm_kernel<<<B*S, 256, 0, stream>>>(tv, tvnorm, D);

  // 9) wvn output
  wvn_kernel<<<B*(T/32), 128, 0, stream>>>(
      q2, D, kv2, 2*D, 0, m_buf, l_buf, tvnorm, out_wvn, B, H, T, S);

  // 10) MLP: out = x2 + gelu(LN(x2) @ w1^T + b1) @ w2^T + b2
  ln_kernel<<<M, 256, 0, stream>>>(x2_f32, ln_g, ln_b, lnx_b, D);
  gemm_bt_kernel<<<dim3(4*D/128, M/128), 256, 0, stream>>>(
      lnx_b, mlp_w1b, mlp_b1, nullptr, nullptr, h_b, M, 4*D, D, 1);
  gemm_bt_kernel<<<dim3(D/128, M/128), 256, 0, stream>>>(
      h_b, mlp_w2b, mlp_b2, x2_f32, out_x, nullptr, M, D, 4*D, 0);
}
